// BornStage_11227044511980
// MI455X (gfx1250) — matrix operand order repaired and checked by reading, not compiled, not run on hardware
//
#include <hip/hip_runtime.h>
#include <hip/hip_bf16.h>

typedef __attribute__((ext_vector_type(16))) __bf16 v16bf;
typedef __attribute__((ext_vector_type(8)))  float  v8f;
typedef __attribute__((ext_vector_type(4)))  unsigned int u32x4;
typedef __attribute__((ext_vector_type(8)))  int i32x8;
typedef __attribute__((ext_vector_type(4)))  int i32x4;

#define BB   8
#define HH   256
#define WWD  256
#define CCH  32
#define NPIX (BB*HH*WWD)      // 524288 pixels
#define NROW (BB*HH)          // 2048 (b,h) rows

#if defined(__has_builtin)
#if __has_builtin(__builtin_amdgcn_tensor_load_to_lds)
#define HAVE_TDM 1
#endif
#endif

// ---------------- helpers ----------------

__device__ __forceinline__ float geluf(float x){
  const float c0 = 0.7978845608028654f, c1 = 0.044715f;
  return 0.5f*x*(1.0f + tanhf(c0*(x + c1*x*x*x)));
}

__device__ __forceinline__ v8f wmma_bf(v16bf a, v16bf b, v8f c){
  // D = A(16x32 bf16) * B(32x16 bf16) + C(16x16 f32)
  return __builtin_amdgcn_wmma_f32_16x16x32_bf16(false, a, false, b, (short)0, c, false, false);
}

__device__ __forceinline__ void lds_fence(){
  asm volatile("s_wait_dscnt 0" ::: "memory");
}

#ifdef HAVE_TDM
// One-wave 1-D DMA: copy nel bf16 elements from global to LDS byte offset lds_off.
// D# per CDNA5 ISA ch.8: group0 = {flags, lds_addr, global_addr, type=2},
// group1 = {data_size=2B, tensor_dim0=nel, tile_dim0=nel, dim1=1, stride0=nel}.
__device__ __forceinline__ void tdm_load_to_lds(const __bf16* gsrc, unsigned lds_off, unsigned nel){
  unsigned long ga = (unsigned long)(const void*)gsrc;
  u32x4 g0;
  g0[0] = 1u;                                               // count=1 (valid user D#)
  g0[1] = lds_off;                                          // lds_addr (bytes)
  g0[2] = (unsigned)(ga & 0xFFFFFFFFu);                     // global_addr[31:0]
  g0[3] = (unsigned)((ga >> 32) & 0x01FFFFFFu) | (2u << 30);// global_addr[56:32] | type=2
  i32x8 g1;
  g1[0] = (int)(1u << 16);                                  // data_size=1 -> 2 bytes
  g1[1] = (int)((nel & 0xFFFFu) << 16);                     // tensor_dim0[15:0]
  g1[2] = (int)(((nel >> 16) & 0xFFFFu) | (1u << 16));      // tensor_dim0[31:16] | tensor_dim1=1
  g1[3] = (int)((nel & 0xFFFFu) << 16);                     // tile_dim0 = nel
  g1[4] = 1;                                                // tile_dim1=1, tile_dim2=0
  g1[5] = (int)nel;                                         // tensor_dim0_stride[31:0]
  g1[6] = 0;
  g1[7] = 0;
  i32x4 z4 = {0,0,0,0};
#if __has_include(<hip/amd_detail/amd_gfx1250_TDM.h>)
  i32x8 z8 = {0,0,0,0,0,0,0,0};
  __builtin_amdgcn_tensor_load_to_lds(g0, g1, z4, z4, z8, 0);
#else
  __builtin_amdgcn_tensor_load_to_lds(g0, g1, z4, z4, 0);
#endif
}
#endif

// K-index map for 16-bit A fragment (16x32): element e, lane-half hlf
__device__ __forceinline__ int a_kmap(int e, int hlf){ return hlf*16 + e; }

// B fragment (32x16) from f32 weight matrix W[kin x cout]; zero-pads k >= kin
__device__ __forceinline__ v16bf load_Bfrag(const float* Wt, int kin, int cout,
                                            int ktile, int ntile, int lane){
  int n = ntile*16 + (lane & 15);
  int kbase = ktile*32 + ((lane >> 4) * 16);
  v16bf f;
#pragma unroll
  for (int e = 0; e < 16; ++e){
    int k = kbase + e;
    f[e] = (__bf16)((k < kin) ? Wt[(long)k*cout + n] : 0.0f);
  }
  return f;
}

// A fragment (16x32) from global f32, rows = pixels, zero-pads k >= kin
__device__ __forceinline__ v16bf load_Afrag_g(const float* X, long rowbase, int stride,
                                              int kin, int lane){
  int row = lane & 15, hlf = lane >> 4;
  const float* p = X + (rowbase + row) * (long)stride;
  v16bf f;
#pragma unroll
  for (int e = 0; e < 16; ++e){
    int k = a_kmap(e, hlf);
    f[e] = (__bf16)((k < kin) ? p[k] : 0.0f);
  }
  return f;
}

// A fragment (16x32) gathered from per-wave LDS tile [16 x stride] f32
__device__ __forceinline__ v16bf load_Afrag_lds(const float* l, int stride, int koff, int lane){
  int row = lane & 15, hlf = lane >> 4;
  v16bf f;
#pragma unroll
  for (int e = 0; e < 16; ++e) f[e] = (__bf16)l[row*stride + koff + a_kmap(e, hlf)];
  return f;
}

// store a 16x16 f32 D tile into LDS [16 x stride]
__device__ __forceinline__ void store_D_lds(float* l, int stride, int ntile, v8f d, int lane){
  int n = ntile*16 + (lane & 15); int hlf = lane >> 4;
#pragma unroll
  for (int r = 0; r < 8; ++r) l[(r + 8*hlf)*stride + n] = d[r];
}

// ---------------- kernel 0: twiddle fragment tables ----------------
// Fw: forward DFT-W A-frags [mtile(2)][ktile(8)][lane(32)][e(16)] bf16
// Iw: inverse DFT-W A-frags [wtile(16)][lane(32)][e(16)]       bf16 (norm 1/65536, alpha_j folded)
__global__ void __launch_bounds__(512) init_tables_kernel(__bf16* Fw, __bf16* Iw){
  int idx = blockIdx.x*blockDim.x + threadIdx.x;
  if (idx >= 8192) return;
  const float TP = 6.283185307179586f;
  {
    int e = idx & 15, lane = (idx >> 4) & 31, kt = (idx >> 9) & 7, mt = idx >> 12;
    int row = lane & 15, hlf = lane >> 4;
    int m = mt*16 + row;                            // mode row: 2j+ri
    int w = kt*32 + a_kmap(e, hlf);                 // spatial index
    float val = 0.0f;
    if (m < 24){
      int j = m >> 1, ri = m & 1;
      float th = TP * (float)((j*w) & 255) / 256.0f;
      val = ri ? -__sinf(th) : __cosf(th);          // e^{-i th}
    }
    Fw[idx] = (__bf16)val;
  }
  {
    int e = idx & 15, lane = (idx >> 4) & 31, wt = idx >> 9;
    int row = lane & 15, hlf = lane >> 4;
    int w = wt*16 + row;
    int k = a_kmap(e, hlf);                         // jri index, K padded to 32
    float val = 0.0f;
    if (k < 24){
      int j = k >> 1, ri = k & 1;
      float amp = ((j == 0) ? 1.0f : 2.0f) * (1.0f/65536.0f);
      float th = TP * (float)((j*w) & 255) / 256.0f;
      val = amp * (ri ? -__sinf(th) : __cosf(th));  // Re(T e^{+i th})
    }
    Iw[idx] = (__bf16)val;
  }
}

// ---------------- kernel 1: k-path (L, G projections + Ms) ----------------
// proj loop is OUTER so only one projection's weight fragments are live at a time
__global__ void __launch_bounds__(128) __attribute__((amdgpu_waves_per_eu(4)))
kpath_kernel(
    const float* __restrict__ kf,
    const float* Lw1, const float* Lb1, const float* Lw2, const float* Lb2,
    const float* Gw1, const float* Gb1, const float* Gw2, const float* Gb2,
    const float* Msw, const float* Msbp,
    __bf16* __restrict__ Lbuf, __bf16* __restrict__ Gbuf, __bf16* __restrict__ Msbuf)
{
  __shared__ float lds[4][16*64];
  int lane = threadIdx.x & 31;
  int wv = threadIdx.x >> 5;
  int wave = blockIdx.x*4 + wv;        // 0..2047
  float* myl = lds[wv];
  int hlf = lane >> 4, col = lane & 15;

#pragma unroll
  for (int proj = 0; proj < 2; ++proj){
    const float* w1 = proj ? Gw1 : Lw1;
    const float* b1p = proj ? Gb1 : Lb1;
    const float* w2 = proj ? Gw2 : Lw2;
    const float* b2p = proj ? Gb2 : Lb2;
    __bf16* dst = proj ? Gbuf : Lbuf;

    v16bf B1[4], B2[4];
    float b1[4], b2[2];
#pragma unroll
    for (int nt = 0; nt < 4; ++nt){ B1[nt] = load_Bfrag(w1, 4, 64, 0, nt, lane); b1[nt] = b1p[nt*16+col]; }
#pragma unroll
    for (int kt = 0; kt < 2; ++kt)
#pragma unroll
      for (int nt = 0; nt < 2; ++nt) B2[kt*2+nt] = load_Bfrag(w2, 64, 32, kt, nt, lane);
#pragma unroll
    for (int nt = 0; nt < 2; ++nt) b2[nt] = b2p[nt*16+col];

    for (int t = 0; t < 16; ++t){
      int tile = wave + t*2048;
      long rowbase = (long)tile * 16;
      if (t < 15) __builtin_prefetch(kf + (rowbase + 2048*16 + (lane & 15))*4, 0, 1);
      v16bf Ak = load_Afrag_g(kf, rowbase, 4, 4, lane);
      v8f h[4];
#pragma unroll
      for (int nt = 0; nt < 4; ++nt){
        v8f z = {};
        h[nt] = wmma_bf(Ak, B1[nt], z);
#pragma unroll
        for (int r = 0; r < 8; ++r) h[nt][r] = geluf(h[nt][r] + b1[nt]);
        store_D_lds(myl, 64, nt, h[nt], lane);
      }
      lds_fence();
      v16bf Ah0 = load_Afrag_lds(myl, 64, 0, lane);
      v16bf Ah1 = load_Afrag_lds(myl, 64, 32, lane);
      lds_fence();
#pragma unroll
      for (int nt = 0; nt < 2; ++nt){
        v8f z = {};
        v8f o = wmma_bf(Ah0, B2[nt], z);
        o = wmma_bf(Ah1, B2[2+nt], o);
        int c = nt*16 + col;
#pragma unroll
        for (int r = 0; r < 8; ++r)
          dst[(rowbase + r + 8*hlf)*32 + c] = (__bf16)(o[r] + b2[nt]);
      }
    }
  }

  // Ms = k @ Ms_w + Ms_b
  {
    v16bf MsB[2]; float mb[2];
#pragma unroll
    for (int nt = 0; nt < 2; ++nt){ MsB[nt] = load_Bfrag(Msw, 4, 32, 0, nt, lane); mb[nt] = Msbp[nt*16+col]; }
    for (int t = 0; t < 16; ++t){
      int tile = wave + t*2048;
      long rowbase = (long)tile * 16;
      v16bf Ak = load_Afrag_g(kf, rowbase, 4, 4, lane);
#pragma unroll
      for (int nt = 0; nt < 2; ++nt){
        v8f z = {};
        v8f o = wmma_bf(Ak, MsB[nt], z);
        int c = nt*16 + col;
#pragma unroll
        for (int r = 0; r < 8; ++r)
          Msbuf[(rowbase + r + 8*hlf)*32 + c] = (__bf16)(o[r] + mb[nt]);
      }
    }
  }
}

// ---------------- kernel 2: x-path (u, Wu, Hu, v = L*u + Ms) ----------------
__global__ void __launch_bounds__(128) __attribute__((amdgpu_waves_per_eu(4)))
xpath_kernel(
    const float* __restrict__ x,
    const float* uw,  const float* ubp,
    const float* Wuw, const float* Wubp,
    const float* Huw, const float* Hubp,
    const __bf16* __restrict__ Lbuf, const __bf16* __restrict__ Msbuf,
    __bf16* __restrict__ WuOut, __bf16* __restrict__ HuOut, __bf16* __restrict__ vt)
{
  __shared__ float lds[4][16*32];
  int lane = threadIdx.x & 31;
  int wv = threadIdx.x >> 5;
  int wave = blockIdx.x*4 + wv;        // 0..2047
  float* myl = lds[wv];
  int hlf = lane >> 4, col = lane & 15;

  v16bf uB[2], WuB[2], HuB[2];
  float ubv[2], Wubv[2], Hubv[2];
#pragma unroll
  for (int nt = 0; nt < 2; ++nt){
    uB[nt]  = load_Bfrag(uw, 32, 32, 0, nt, lane);
    WuB[nt] = load_Bfrag(Wuw, 32, 32, 0, nt, lane);
    HuB[nt] = load_Bfrag(Huw, 32, 32, 0, nt, lane);
    ubv[nt] = ubp[nt*16+col]; Wubv[nt] = Wubp[nt*16+col]; Hubv[nt] = Hubp[nt*16+col];
  }

  for (int t = 0; t < 16; ++t){
    int tile = wave + t*2048;
    long rowbase = (long)tile * 16;
    if (t < 15) __builtin_prefetch(x + (rowbase + 2048*16 + (lane & 15))*32, 0, 1);
    v16bf Ax = load_Afrag_g(x, rowbase, 32, 32, lane);

    v8f uD[2];
#pragma unroll
    for (int nt = 0; nt < 2; ++nt){
      v8f z = {};
      uD[nt] = wmma_bf(Ax, uB[nt], z);
#pragma unroll
      for (int r = 0; r < 8; ++r) uD[nt][r] += ubv[nt];
      store_D_lds(myl, 32, nt, uD[nt], lane);
    }
    lds_fence();
    v16bf Au = load_Afrag_lds(myl, 32, 0, lane);
    lds_fence();

    int bh = tile >> 4, wt = tile & 15;
#pragma unroll
    for (int nt = 0; nt < 2; ++nt){
      int c = nt*16 + col;
      v8f z = {};
      v8f wuD = wmma_bf(Au, WuB[nt], z);
      v8f huD = wmma_bf(Au, HuB[nt], z);
#pragma unroll
      for (int r = 0; r < 8; ++r){
        int m = r + 8*hlf;
        long pix = rowbase + m;
        WuOut[pix*32 + c] = (__bf16)(wuD[r] + Wubv[nt]);
        HuOut[pix*32 + c] = (__bf16)(huD[r] + Hubv[nt]);
        float v = (float)Lbuf[pix*32 + c]*uD[nt][r] + (float)Msbuf[pix*32 + c];
        vt[((long)bh*32 + c)*256 + wt*16 + m] = (__bf16)v;   // [B,H,C,W] layout
      }
    }
  }
}

// ---------------- kernel 3: forward truncated DFT along W (GEMM) ----------------
// S1[b,h, m=2j+ri (24), c] = Re/Im of sum_w v[b,h,w,c] e^{-2pi i j w/256}
// The 16KB v-row per wave is staged into a private LDS slice by the Tensor Data
// Mover (one DMA per wave, waited with s_wait_tensorcnt), then consumed as
// contiguous ds_load_b128 B-fragments feeding the WMMA pipe.
__global__ void __launch_bounds__(128) __attribute__((amdgpu_waves_per_eu(4)))
dftw_kernel(const __bf16* __restrict__ vt,
            const __bf16* __restrict__ Fw,
            float* __restrict__ S1)
{
#ifdef HAVE_TDM
  __shared__ __bf16 vlds[4][32*256];           // 16KB per wave
#endif
  int lane = threadIdx.x & 31;
  int wv = threadIdx.x >> 5;
  int bh = blockIdx.x*4 + wv;                  // 0..2047
  int hlf = lane >> 4, col = lane & 15;

#ifdef HAVE_TDM
  tdm_load_to_lds(vt + (long)bh*8192, (unsigned)wv*(32*256*2), 8192);
  __builtin_amdgcn_s_wait_tensorcnt(0);
  const __bf16* vrow = &vlds[wv][0];
#else
  const __bf16* vrow = vt + (long)bh * 32 * 256;
#endif

  v8f acc[2][2] = {};
#pragma unroll
  for (int kt = 0; kt < 8; ++kt){
    v16bf Bf[2];
#pragma unroll
    for (int nt = 0; nt < 2; ++nt){
      int c = nt*16 + col;
      Bf[nt] = *(const v16bf*)(vrow + (long)c*256 + kt*32 + hlf*16);  // contiguous bf16
    }
#pragma unroll
    for (int mt = 0; mt < 2; ++mt){
      v16bf Af = *(const v16bf*)(Fw + ((long)(mt*8 + kt)*32 + lane)*16);
      acc[mt][0] = wmma_bf(Af, Bf[0], acc[mt][0]);
      acc[mt][1] = wmma_bf(Af, Bf[1], acc[mt][1]);
    }
  }
#pragma unroll
  for (int mt = 0; mt < 2; ++mt)
#pragma unroll
    for (int nt = 0; nt < 2; ++nt)
#pragma unroll
      for (int r = 0; r < 8; ++r){
        int m = mt*16 + r + 8*hlf;
        if (m < 24) S1[((long)bh*24 + m)*32 + nt*16 + col] = acc[mt][nt][r];
      }
}

// ---------------- kernel 4: DFT along H + spectral mode mixing ----------------
__global__ void __launch_bounds__(32) modemix_kernel(const float* __restrict__ S1,
    const float* __restrict__ k1r, const float* __restrict__ k1i,
    const float* __restrict__ k2r, const float* __restrict__ k2i,
    float* __restrict__ S3)
{
  __shared__ float xr[32], xi[32];
  int c = threadIdx.x;
  int id = blockIdx.x;                 // b*288 + ii*12 + j
  int b = id / 288; int rem = id - b*288;
  int ii = rem / 12; int j = rem - ii*12;
  int ieff = (ii < 12) ? ii : ii - 24;
  float w0 = 6.283185307179586f * (float)ieff / 256.0f;
  float re = 0.f, im = 0.f;
  const float* base = S1 + (long)b*256*24*32 + (long)(2*j)*32 + c;
  for (int h = 0; h < 256; ++h){
    float th = w0 * (float)h;
    float s, cs; __sincosf(th, &s, &cs);
    float vr = base[(long)h*24*32];
    float vi = base[(long)h*24*32 + 32];
    re += vr*cs + vi*s;                // (Re + i Im) * e^{-i th}
    im += vi*cs - vr*s;
  }
  xr[c] = re; xi[c] = im;
  __syncthreads();
  const float* Kr = (ii < 12) ? k1r : k2r;
  const float* Ki = (ii < 12) ? k1i : k2i;
  int im_ = (ii < 12) ? ii : ii - 12;
  float ar = 0.f, ai = 0.f;
  for (int cc = 0; cc < 32; ++cc){
    long ko = (((long)cc*32 + c)*12 + im_)*12 + j;   // K[c,o,i,j], o = this thread
    float kr = Kr[ko], ki = Ki[ko];
    ar += xr[cc]*kr - xi[cc]*ki;
    ai += xr[cc]*ki + xi[cc]*kr;
  }
  long ob = (long)id * 64;
  S3[ob + c] = ar;
  S3[ob + 32 + c] = ai;
}

// ---------------- kernel 5: inverse DFT along H ----------------
__global__ void __launch_bounds__(32) idfth_kernel(const float* __restrict__ S3,
                                                   float* __restrict__ T)
{
  int o = threadIdx.x;
  int id = blockIdx.x;                 // b*3072 + h*12 + j
  int b = id / 3072; int rem = id - b*3072;
  int h = rem / 12; int j = rem - h*12;
  float tr = 0.f, ti = 0.f;
  for (int ii = 0; ii < 24; ++ii){
    int ieff = (ii < 12) ? ii : ii - 24;
    float th = 6.283185307179586f * (float)(ieff*h) / 256.0f;
    float s, cs; __sincosf(th, &s, &cs);
    long sb = (((long)b*24 + ii)*12 + j)*64;
    float sr = S3[sb + o], si = S3[sb + 32 + o];
    tr += sr*cs - si*s;                // (Re + i Im) * e^{+i th}
    ti += sr*s + si*cs;
  }
  long tb = (((long)b*256 + h)*24 + 2*j)*32;
  T[tb + o] = tr;
  T[tb + 32 + o] = ti;
}

// ---------------- kernel 6: inverse DFT-W (WMMA) + born + out GEMM + gelu ----------------
__global__ void __launch_bounds__(128) __attribute__((amdgpu_waves_per_eu(4)))
final_kernel(const float* __restrict__ T,
    const __bf16* __restrict__ Iw,
    const __bf16* __restrict__ Wu, const __bf16* __restrict__ G,
    const __bf16* __restrict__ Hu,
    const float* __restrict__ outw, const float* __restrict__ outb,
    float* __restrict__ out)
{
  __shared__ float lds[4][16*32];
  int lane = threadIdx.x & 31, wv = threadIdx.x >> 5;
  int bh = blockIdx.x*4 + wv;          // 0..2047
  float* myl = lds[wv];
  int hlf = lane >> 4, col = lane & 15;

  // B fragments from T[b,h] (24x32 padded to 32), reused for all 16 w-tiles
  v16bf TB[2];
#pragma unroll
  for (int nt = 0; nt < 2; ++nt){
    int c = nt*16 + col;
#pragma unroll
    for (int e = 0; e < 16; ++e){
      int k = hlf*16 + e;
      TB[nt][e] = (__bf16)((k < 24) ? T[((long)bh*24 + k)*32 + c] : 0.0f);
    }
  }
  v16bf OW[2]; float obv[2];
#pragma unroll
  for (int nt = 0; nt < 2; ++nt){
    OW[nt] = load_Bfrag(outw, 32, 32, 0, nt, lane);
    obv[nt] = outb[nt*16 + col];
  }

  for (int wt = 0; wt < 16; ++wt){
    v16bf Af = *(const v16bf*)(Iw + ((long)wt*32 + lane)*16);
    v8f fv[2], born[2];
#pragma unroll
    for (int nt = 0; nt < 2; ++nt){ v8f z = {}; fv[nt] = wmma_bf(Af, TB[nt], z); }
    long pixbase = (long)bh*256 + wt*16;
#pragma unroll
    for (int nt = 0; nt < 2; ++nt){
      int c = nt*16 + col;
#pragma unroll
      for (int r = 0; r < 8; ++r){
        long a = (pixbase + r + 8*hlf)*32 + c;
        float wu = (float)Wu[a], g = (float)G[a], hu = (float)Hu[a];
        born[nt][r] = wu - g*(hu - fv[nt][r]);
      }
      store_D_lds(myl, 32, nt, born[nt], lane);
    }
    lds_fence();
    v16bf Ab = load_Afrag_lds(myl, 32, 0, lane);
    lds_fence();
#pragma unroll
    for (int nt = 0; nt < 2; ++nt){
      v8f z = {};
      v8f od = wmma_bf(Ab, OW[nt], z);
#pragma unroll
      for (int r = 0; r < 8; ++r)
        out[(pixbase + r + 8*hlf)*32 + nt*16 + col] = geluf(od[r] + obv[nt]);
    }
  }
}

// ---------------- launch ----------------
extern "C" void kernel_launch(void* const* d_in, const int* in_sizes, int n_in,
                              void* d_out, int out_size, void* d_ws, size_t ws_size,
                              hipStream_t stream)
{
  (void)in_sizes; (void)n_in; (void)out_size; (void)ws_size;
  const float* x    = (const float*)d_in[0];
  const float* kf   = (const float*)d_in[1];
  const float* Lw1  = (const float*)d_in[2];  const float* Lb1 = (const float*)d_in[3];
  const float* Lw2  = (const float*)d_in[4];  const float* Lb2 = (const float*)d_in[5];
  const float* Gw1  = (const float*)d_in[6];  const float* Gb1 = (const float*)d_in[7];
  const float* Gw2  = (const float*)d_in[8];  const float* Gb2 = (const float*)d_in[9];
  const float* uw   = (const float*)d_in[10]; const float* ub  = (const float*)d_in[11];
  const float* Wuw  = (const float*)d_in[12]; const float* Wub = (const float*)d_in[13];
  const float* Huw  = (const float*)d_in[14]; const float* Hub = (const float*)d_in[15];
  const float* Msw  = (const float*)d_in[16]; const float* Msb = (const float*)d_in[17];
  const float* outw = (const float*)d_in[18]; const float* outb= (const float*)d_in[19];
  const float* k1r  = (const float*)d_in[20]; const float* k1i = (const float*)d_in[21];
  const float* k2r  = (const float*)d_in[22]; const float* k2i = (const float*)d_in[23];
  float* out = (float*)d_out;

  char* wp = (char*)d_ws;
  auto take = [&](size_t bytes){ char* p = wp; wp += (bytes + 255) & ~(size_t)255; return p; };
  __bf16* Lbuf  = (__bf16*)take((size_t)NPIX*32*2);
  __bf16* Gbuf  = (__bf16*)take((size_t)NPIX*32*2);
  __bf16* Msbuf = (__bf16*)take((size_t)NPIX*32*2);
  __bf16* Wubuf = (__bf16*)take((size_t)NPIX*32*2);
  __bf16* Hubuf = (__bf16*)take((size_t)NPIX*32*2);
  __bf16* vt    = (__bf16*)take((size_t)NPIX*32*2);
  float*  S1    = (float*) take((size_t)NROW*24*32*4);
  float*  S3    = (float*) take((size_t)BB*24*12*2*32*4);
  float*  T     = (float*) take((size_t)NROW*24*32*4);
  __bf16* Fw    = (__bf16*)take(8192*2);
  __bf16* Iw    = (__bf16*)take(8192*2);

  init_tables_kernel<<<16, 512, 0, stream>>>(Fw, Iw);
  kpath_kernel<<<512, 128, 0, stream>>>(kf,
      Lw1, Lb1, Lw2, Lb2, Gw1, Gb1, Gw2, Gb2, Msw, Msb,
      Lbuf, Gbuf, Msbuf);
  xpath_kernel<<<512, 128, 0, stream>>>(x,
      uw, ub, Wuw, Wub, Huw, Hub,
      Lbuf, Msbuf, Wubuf, Hubuf, vt);
  dftw_kernel<<<512, 128, 0, stream>>>(vt, Fw, S1);
  modemix_kernel<<<BB*288, 32, 0, stream>>>(S1, k1r, k1i, k2r, k2i, S3);
  idfth_kernel<<<BB*3072, 32, 0, stream>>>(S3, T);
  final_kernel<<<512, 128, 0, stream>>>(T, Iw, Wubuf, Gbuf, Hubuf, outw, outb, out);
}
